// Model_88055419503292
// MI455X (gfx1250) — compile-verified, matrix operand order checked by reading, not run on hardware
//
#include <hip/hip_runtime.h>
#include <hip/hip_bf16.h>
#include <cmath>

// ---------------------------------------------------------------------------
// Types / WMMA helpers (gfx1250 wave32 WMMA f32 = A(16x32 f16) x B(32x16 f16))
// ---------------------------------------------------------------------------
typedef __attribute__((ext_vector_type(16))) _Float16     v16h;
typedef __attribute__((ext_vector_type(8)))  float        v8f;
typedef __attribute__((ext_vector_type(4)))  unsigned int v4u;
typedef __attribute__((ext_vector_type(8)))  int          v8i;
typedef __attribute__((ext_vector_type(4)))  int          v4i;

__device__ __forceinline__ v8f vzero8() {
  v8f z = {0.f, 0.f, 0.f, 0.f, 0.f, 0.f, 0.f, 0.f};
  return z;
}

__device__ __forceinline__ v8f wmma_f16(v16h a, v16h b, v8f c) {
  // (neg_a, A, neg_b, B, c_mod, C, reuse_a, reuse_b)
  return __builtin_amdgcn_wmma_f32_16x16x32_f16(false, a, false, b, (short)0, c,
                                                false, false);
}

// A-fragment loader: 16x32 (MxK) f16 tile, row-major in LDS with leading dim ld.
// Per ISA 7.12.2: lane L holds row M=L%16; halves map to K as:
//   vg = h/2, pos = h%2, K = (vg>=4?16:0) + (L>=16?8:0) + (vg%4)*2 + pos
__device__ __forceinline__ v16h load_frag_a_lds(const _Float16* base, int ld) {
  int lane = threadIdx.x & 31;
  int m = lane & 15;
  int khi = (lane >> 4) << 3;
  v16h a;
#pragma unroll
  for (int h = 0; h < 16; ++h) {
    int vg = h >> 1;
    int k = ((vg & 4) << 2) + khi + ((vg & 3) << 1) + (h & 1);
    a[h] = base[m * ld + k];
  }
  return a;
}

// B fragments are pre-swizzled into [n_tile][k_chunk][lane] v16h records, so a
// fragment load is one contiguous 32B/lane vector load (2x global_load_b128).
__device__ __forceinline__ v16h load_packed_b(const _Float16* __restrict__ packed,
                                              int n_tile, int k_chunk, int k_chunks) {
  int lane = threadIdx.x & 31;
  const v16h* p = (const v16h*)packed;
  return p[((size_t)n_tile * k_chunks + k_chunk) * 32 + lane];
}

__device__ __forceinline__ float sigmoidf_(float x) {
  return 1.f / (1.f + expf(-x));
}

// ---------------------------------------------------------------------------
// Problem constants
// ---------------------------------------------------------------------------
#define N_ORDERS 200000
#define N_BATCH  8192
#define MAX_LEN  32
#define EDIM     128
#define HISTN    16
#define DIM_B    532   // 4*E + 3 + 1 + 16
#define DIM_BP   544   // padded to 17 * 32 for WMMA K
#define GM       32    // batch rows per GRU/head block

// ---------------------------------------------------------------------------
// Kernel 0a: repack fp32 weight matrix W[n_out][k_in] into f16 WMMA B-fragments
// layout [n_tiles][k_chunks][32 lanes][16 halves]; zero-pads K beyond k_in.
// ---------------------------------------------------------------------------
__global__ void pack_weights(const float* __restrict__ W, _Float16* __restrict__ dst,
                             int n_out, int k_in, int n_tiles, int k_chunks) {
  int idx = blockIdx.x * blockDim.x + threadIdx.x;  // one per (n_tile,k_chunk,lane)
  int total = n_tiles * k_chunks * 32;
  if (idx >= total) return;
  int lane = idx & 31;
  int chunk = (idx >> 5) % k_chunks;
  int ntile = (idx >> 5) / k_chunks;
  int n = ntile * 16 + (lane & 15);
  int khi = (lane >> 4) << 3;
  v16h v;
#pragma unroll
  for (int h = 0; h < 16; ++h) {
    int vg = h >> 1;
    int k = chunk * 32 + ((vg & 4) << 2) + khi + ((vg & 3) << 1) + (h & 1);
    float x = (k < k_in && n < n_out) ? W[(size_t)n * k_in + k] : 0.f;
    v[h] = (_Float16)x;
  }
  ((v16h*)dst)[idx] = v;
}

__global__ void zero128(float* __restrict__ p) { p[threadIdx.x] = 0.f; }

// ---------------------------------------------------------------------------
// Kernel 1: order embedding.  128 orders / block (8 waves x 16-row M tiles).
// ---------------------------------------------------------------------------
__global__ __launch_bounds__(256) void order_embed(
    const float* __restrict__ order_usage,
    const float* __restrict__ ln_g, const float* __restrict__ ln_b,
    const float* __restrict__ w1, const float* __restrict__ b1,
    const _Float16* __restrict__ pk_w2, const float* __restrict__ b2,
    _Float16* __restrict__ o3h) {
  __shared__ float    o2S[128][4];
  __shared__ _Float16 actA[128][264];  // 128 x 256 (+8 pad)

  int tid = threadIdx.x;
  int lane = tid & 31, wv = tid >> 5;
  int obase = blockIdx.x * 128;

  if (tid < 128) {
    int o = obase + tid;
    float x0 = 0.f, x1 = 0.f, x2 = 0.f;
    if (o < N_ORDERS) {
      x0 = order_usage[o * 3 + 0];
      x1 = order_usage[o * 3 + 1];
      x2 = order_usage[o * 3 + 2];
    }
    float m = (x0 + x1 + x2) * (1.f / 3.f);
    float d0 = x0 - m, d1 = x1 - m, d2 = x2 - m;
    float var = (d0 * d0 + d1 * d1 + d2 * d2) * (1.f / 3.f);
    float rs = rsqrtf(var + 1e-5f);
    o2S[tid][0] = d0 * rs * ln_g[0] + ln_b[0];
    o2S[tid][1] = d1 * rs * ln_g[1] + ln_b[1];
    o2S[tid][2] = d2 * rs * ln_g[2] + ln_b[2];
  }
  __syncthreads();

  {  // layer 1 (K=3): each thread fills 128 (order, channel) cells
    int ol = tid & 127;
    int cb = tid >> 7;
    float y0 = o2S[ol][0], y1 = o2S[ol][1], y2 = o2S[ol][2];
#pragma unroll 4
    for (int i = 0; i < 128; ++i) {
      int c = cb + 2 * i;
      float a = y0 * w1[c * 3] + y1 * w1[c * 3 + 1] + y2 * w1[c * 3 + 2] + b1[c];
      actA[ol][c] = (_Float16)tanhf(a);
    }
  }
  __syncthreads();

  {  // layer 2: wave wv owns 16 orders, 8 N tiles x 8 K chunks = 64 WMMAs
    v8f acc[8];
#pragma unroll
    for (int nt = 0; nt < 8; ++nt) acc[nt] = vzero8();
#pragma unroll
    for (int kc = 0; kc < 8; ++kc) {
      v16h a = load_frag_a_lds(&actA[wv * 16][kc * 32], 264);
#pragma unroll
      for (int nt = 0; nt < 8; ++nt)
        acc[nt] = wmma_f16(a, load_packed_b(pk_w2, nt, kc, 8), acc[nt]);
    }
    int n15 = lane & 15;
    int mb = (lane >> 4) << 3;
#pragma unroll
    for (int nt = 0; nt < 8; ++nt) {
      float bias = b2[nt * 16 + n15];
#pragma unroll
      for (int r = 0; r < 8; ++r) {
        int o = obase + wv * 16 + mb + r;
        if (o < N_ORDERS)
          o3h[(size_t)o * EDIM + nt * 16 + n15] = (_Float16)(acc[nt][r] + bias);
      }
    }
  }
}

// ---------------------------------------------------------------------------
// Kernel 2: GRU + masked pooling.  32 batch rows / block, 256 blocks.
// Software-pipelined gather (x_{t+1} loads hide behind step-t WMMAs) and only
// 2 barriers/step: gate-update only reads rzS/innS/hnS, next phase-A only
// writes hA/xA, so they fuse without a barrier.
// ---------------------------------------------------------------------------
__global__ __launch_bounds__(256) void gru_kernel(
    const _Float16* __restrict__ o3h,
    const int* __restrict__ plan_idx, const int* __restrict__ plan_len,
    const _Float16* __restrict__ pk_wih, const _Float16* __restrict__ pk_whh,
    const float* __restrict__ bih, const float* __restrict__ bhh,
    const float* __restrict__ hist, const float* __restrict__ plan_usage,
    float* __restrict__ b2in) {
  __shared__ _Float16 hA[GM][136];
  __shared__ _Float16 xA[GM][136];
  __shared__ float    rzS[GM][256];   // r gate cols 0..127, z gate cols 128..255
  __shared__ float    innS[GM][128];
  __shared__ float    hnS[GM][128];

  int tid = threadIdx.x;
  int lane = tid & 31, wv = tid >> 5;
  int mbase = blockIdx.x * GM;
  int col = tid & 127;        // owned E-column
  int mo = (tid >> 7) * 16;   // owned row group (0 or 16)
  int mt = wv >> 2, g = wv & 3;

  // Warm the packed recurrent weights into cache (reused 32x per block).
  for (int off = tid * 256; off < 24 * 4 * 1024; off += 256 * 256) {
    __builtin_prefetch((const char*)pk_wih + off, 0, 3);
    __builtin_prefetch((const char*)pk_whh + off, 0, 3);
  }

  float hreg[16], ssum[16], smax[16], smin[16];
  int lens[16];
  _Float16 xh[16];
#pragma unroll
  for (int i = 0; i < 16; ++i) {
    hreg[i] = 0.f; ssum[i] = 0.f; smax[i] = -INFINITY; smin[i] = INFINITY;
    lens[i] = plan_len[mbase + mo + i];
    int idx0 = plan_idx[(size_t)(mbase + mo + i) * MAX_LEN];
    xh[i] = o3h[(size_t)idx0 * EDIM + col];   // t = 0 gather
  }

#pragma unroll 1
  for (int t = 0; t < MAX_LEN; ++t) {
    // phase A: publish h and x_t to LDS, accumulate pooling
#pragma unroll
    for (int i = 0; i < 16; ++i) {
      int m = mo + i;
      hA[m][col] = (_Float16)hreg[i];
      xA[m][col] = xh[i];
      if (t < lens[i]) {
        float xf = (float)xh[i];
        ssum[i] += xf;
        smax[i] = fmaxf(smax[i], xf);
        smin[i] = fminf(smin[i], xf);
      }
    }
    // issue next step's gather now; latency hides behind phase B WMMAs
    _Float16 xnext[16];
    if (t + 1 < MAX_LEN) {
#pragma unroll
      for (int i = 0; i < 16; ++i) {
        int idx = plan_idx[(size_t)(mbase + mo + i) * MAX_LEN + (t + 1)];
        xnext[i] = o3h[(size_t)idx * EDIM + col];
      }
    }
    __syncthreads();  // barrier 1: hA/xA visible to all waves

    // phase B: gi = x@Wih.T, gh = h@Whh.T  (6 N-tiles/wave x 4 K x 2 = 48 WMMA)
    v16h ax[4], ah[4];
#pragma unroll
    for (int kc = 0; kc < 4; ++kc) {
      ax[kc] = load_frag_a_lds(&xA[mt * 16][kc * 32], 136);
      ah[kc] = load_frag_a_lds(&hA[mt * 16][kc * 32], 136);
    }
#pragma unroll
    for (int j = 0; j < 6; ++j) {
      int nt = g * 6 + j;
      v8f gi = vzero8(), gh = vzero8();
#pragma unroll
      for (int kc = 0; kc < 4; ++kc) {
        gi = wmma_f16(ax[kc], load_packed_b(pk_wih, nt, kc, 4), gi);
        gh = wmma_f16(ah[kc], load_packed_b(pk_whh, nt, kc, 4), gh);
      }
      int n15 = lane & 15;
      int nglob = nt * 16 + n15;
      float bi = bih[nglob], bh = bhh[nglob];
      int mb = mt * 16 + ((lane >> 4) << 3);
      if (nt < 16) {  // r / z gates: store gi+gh+biases
#pragma unroll
        for (int r = 0; r < 8; ++r)
          rzS[mb + r][nglob] = gi[r] + gh[r] + bi + bh;
      } else {        // n gate: keep halves separate for r * hn
        int nc = nglob - 256;
#pragma unroll
        for (int r = 0; r < 8; ++r) {
          innS[mb + r][nc] = gi[r] + bi;
          hnS[mb + r][nc]  = gh[r] + bh;
        }
      }
    }
    __syncthreads();  // barrier 2: gate buffers visible

    // phase C: gate math + masked hidden update (registers only; no barrier —
    // next phase A writes only hA/xA which every wave finished reading)
#pragma unroll
    for (int i = 0; i < 16; ++i) {
      int m = mo + i;
      float r = sigmoidf_(rzS[m][col]);
      float z = sigmoidf_(rzS[m][128 + col]);
      float n = tanhf(innS[m][col] + r * hnS[m][col]);
      float hn = (1.f - z) * n + z * hreg[i];
      if (t < lens[i]) hreg[i] = hn;
    }
    if (t + 1 < MAX_LEN) {
#pragma unroll
      for (int i = 0; i < 16; ++i) xh[i] = xnext[i];
    }
  }

  // emit concat row: [hT | s_mean | s_max | s_min | hist | plan_usage | 0...]
#pragma unroll
  for (int i = 0; i < 16; ++i) {
    int row = mbase + mo + i;
    size_t base = (size_t)row * DIM_BP;
    b2in[base + col]        = hreg[i];
    b2in[base + 128 + col]  = ssum[i] / (float)lens[i];
    b2in[base + 256 + col]  = smax[i];
    b2in[base + 384 + col]  = smin[i];
    if (col < 32) {
      float v = 0.f;
      if (col < HISTN)          v = hist[row * HISTN + col];
      else if (col < HISTN + 4) v = plan_usage[row * 4 + (col - HISTN)];
      b2in[base + 512 + col] = v;  // cols 532..543 zero (WMMA K padding)
    }
  }
}

// ---------------------------------------------------------------------------
// Kernel 3: TDM-stage the 32x544 f32 concat tile into LDS, LN(532),
// tanh(b2 @ eb_w1.T) [K=544] -> @ eb_w2.T [K=256], atomic pooled sums.
// The staging buffer aliases the later act1h/b3S buffers (disjoint lifetimes).
// ---------------------------------------------------------------------------
__global__ __launch_bounds__(256) void head_kernel(
    const float* __restrict__ b2in,
    const float* __restrict__ lnb_g, const float* __restrict__ lnb_b,
    const _Float16* __restrict__ pk_w1, const float* __restrict__ eb_b1,
    const _Float16* __restrict__ pk_w2, const float* __restrict__ eb_b2,
    float* __restrict__ pooled) {
  __shared__ union {
    float b2f[GM][DIM_BP];                         // phase 0-1 (TDM target)
    struct {
      _Float16 act1h[GM][264];
      float    b3S[GM][128];
    } s;                                           // phase 2+
  } u;
  __shared__ _Float16 a2h[GM][DIM_BP];

  int tid = threadIdx.x;
  int lane = tid & 31, wv = tid >> 5;
  int mbase = blockIdx.x * GM;
  int mt = wv >> 2, g = wv & 3;

  if (tid < 32) {  // wave 0 issues the tensor DMA (TDM ignores EXEC; 1 wave only)
    // D# group 0: count=1 | lds_addr | global_addr | type=2 ("image")
    unsigned lds = (unsigned)(uintptr_t)(&u.b2f[0][0]);
    unsigned long long ga =
        (unsigned long long)(uintptr_t)(b2in + (size_t)mbase * DIM_BP);
    v4u g0;
    g0.x = 1u;                                   // count=1, user mode
    g0.y = lds;                                  // LDS byte address
    g0.z = (unsigned)ga;                         // global_addr[31:0]
    g0.w = (unsigned)((ga >> 32) & 0x1FFFFFFu) | (2u << 30);  // [56:32] | type=2
    // D# group 1: data_size=4B, tensor 544x32, tile 544x32, stride0=544
    v8i g1;
    g1[0] = 0x00020000;                          // data_size=2 (4 bytes)
    g1[1] = (int)(544u << 16);                   // tensor_dim0[15:0] @ bits 63:48
    g1[2] = (int)(32u << 16);                    // tensor_dim1[15:0] @ bits 95:80
    g1[3] = (int)(544u << 16);                   // tile_dim0 @ bits 127:112
    g1[4] = 32;                                  // tile_dim1 @ bits 143:128
    g1[5] = 544;                                 // tensor_dim0_stride[31:0]
    g1[6] = 0;
    g1[7] = 0;
    v4i g2 = {0, 0, 0, 0}, g3 = {0, 0, 0, 0};    // 2D tensor: groups 2/3 unused
#if __clang_major__ >= 23
    __builtin_amdgcn_tensor_load_to_lds(g0, g1, g2, g3, g1, 0);
#else
    __builtin_amdgcn_tensor_load_to_lds(g0, g1, g2, g3, 0);
#endif
    __builtin_amdgcn_s_wait_tensorcnt(0);
  }
  __syncthreads();  // staged tile visible to all waves

  __builtin_prefetch((const char*)pk_w1, 0, 3);

  // LayerNorm from LDS: each wave normalizes 4 rows (lane-strided + shfl)
  for (int j = 0; j < 4; ++j) {
    int m = wv * 4 + j;
    float s = 0.f, sq = 0.f;
    for (int k = lane; k < DIM_B; k += 32) {
      float x = u.b2f[m][k];
      s += x; sq += x * x;
    }
#pragma unroll
    for (int o = 16; o > 0; o >>= 1) {
      s  += __shfl_xor(s, o, 32);
      sq += __shfl_xor(sq, o, 32);
    }
    float mean = s * (1.f / (float)DIM_B);
    float var  = sq * (1.f / (float)DIM_B) - mean * mean;
    float rstd = rsqrtf(var + 1e-5f);
    for (int k = lane; k < DIM_BP; k += 32) {
      float v = 0.f;
      if (k < DIM_B)
        v = (u.b2f[m][k] - mean) * rstd * lnb_g[k] + lnb_b[k];
      a2h[m][k] = (_Float16)v;
    }
  }
  __syncthreads();  // a2h ready; b2f dead -> act1h/b3S may reuse the space

  // layer 1: 4 N-tiles/wave x 17 K chunks
#pragma unroll
  for (int j = 0; j < 4; ++j) {
    int nt = g * 4 + j;
    v8f acc = vzero8();
#pragma unroll
    for (int kc = 0; kc < 17; ++kc) {
      v16h a = load_frag_a_lds(&a2h[mt * 16][kc * 32], DIM_BP);
      acc = wmma_f16(a, load_packed_b(pk_w1, nt, kc, 17), acc);
    }
    int n15 = lane & 15;
    float bias = eb_b1[nt * 16 + n15];
    int mb = mt * 16 + ((lane >> 4) << 3);
#pragma unroll
    for (int r = 0; r < 8; ++r)
      u.s.act1h[mb + r][nt * 16 + n15] = (_Float16)tanhf(acc[r] + bias);
  }
  __syncthreads();

  // layer 2: 2 N-tiles/wave x 8 K chunks
#pragma unroll
  for (int j = 0; j < 2; ++j) {
    int nt = g * 2 + j;
    v8f acc = vzero8();
#pragma unroll
    for (int kc = 0; kc < 8; ++kc) {
      v16h a = load_frag_a_lds(&u.s.act1h[mt * 16][kc * 32], 264);
      acc = wmma_f16(a, load_packed_b(pk_w2, nt, kc, 8), acc);
    }
    int n15 = lane & 15;
    float bias = eb_b2[nt * 16 + n15];
    int mb = mt * 16 + ((lane >> 4) << 3);
#pragma unroll
    for (int r = 0; r < 8; ++r)
      u.s.b3S[mb + r][nt * 16 + n15] = acc[r] + bias;
  }
  __syncthreads();

  if (tid < 128) {
    float s = 0.f;
#pragma unroll
    for (int m = 0; m < GM; ++m) s += u.s.b3S[m][tid];
    atomicAdd(&pooled[tid], s);
  }
}

// ---------------------------------------------------------------------------
// Kernel 4: value = tanh(mean(b3) @ cr_w1.T + cr_b1) @ cr_w2.T + cr_b2
// ---------------------------------------------------------------------------
__global__ __launch_bounds__(128) void final_kernel(
    const float* __restrict__ pooled,
    const float* __restrict__ cr_w1, const float* __restrict__ cr_b1,
    const float* __restrict__ cr_w2, const float* __restrict__ cr_b2,
    float* __restrict__ out) {
  __shared__ float pm[128];
  __shared__ float red[128];
  int t = threadIdx.x;
  pm[t] = pooled[t] * (1.f / (float)N_BATCH);
  __syncthreads();
  float acc = cr_b1[t];
  for (int k = 0; k < 128; ++k) acc += pm[k] * cr_w1[t * 128 + k];
  red[t] = tanhf(acc) * cr_w2[t];
  __syncthreads();
  for (int s = 64; s > 0; s >>= 1) {
    if (t < s) red[t] += red[t + s];
    __syncthreads();
  }
  if (t == 0) out[0] = red[0] + cr_b2[0];
}

// ---------------------------------------------------------------------------
// Host launcher
// ---------------------------------------------------------------------------
static inline int cdiv(int a, int b) { return (a + b - 1) / b; }

extern "C" void kernel_launch(void* const* d_in, const int* in_sizes, int n_in,
                              void* d_out, int out_size, void* d_ws, size_t ws_size,
                              hipStream_t stream) {
  // Inputs (setup_inputs dict order)
  const float* order_usage = (const float*)d_in[0];
  const float* hist        = (const float*)d_in[1];
  const float* plan_usage  = (const float*)d_in[2];
  const float* ln_order_g  = (const float*)d_in[3];
  const float* ln_order_b  = (const float*)d_in[4];
  const float* eo_w1       = (const float*)d_in[5];
  const float* eo_b1       = (const float*)d_in[6];
  const float* eo_w2       = (const float*)d_in[7];
  const float* eo_b2       = (const float*)d_in[8];
  const float* gru_wih     = (const float*)d_in[9];
  const float* gru_whh     = (const float*)d_in[10];
  const float* gru_bih     = (const float*)d_in[11];
  const float* gru_bhh     = (const float*)d_in[12];
  const float* lnb_g       = (const float*)d_in[13];
  const float* lnb_b       = (const float*)d_in[14];
  const float* eb_w1       = (const float*)d_in[15];
  const float* eb_b1       = (const float*)d_in[16];
  const float* eb_w2       = (const float*)d_in[17];
  const float* eb_b2       = (const float*)d_in[18];
  const float* cr_w1       = (const float*)d_in[19];
  const float* cr_b1       = (const float*)d_in[20];
  const float* cr_w2       = (const float*)d_in[21];
  const float* cr_b2       = (const float*)d_in[22];
  const int*   plan_idx    = (const int*)d_in[23];
  const int*   plan_len    = (const int*)d_in[24];

  // Workspace layout (bytes); total ~69.7 MB
  char* ws = (char*)d_ws;
  _Float16* pk_eow2 = (_Float16*)(ws + 0);        //  8*8  tiles -> 64 KB
  _Float16* pk_wih  = (_Float16*)(ws + 65536);    // 24*4  tiles -> 96 KB
  _Float16* pk_whh  = (_Float16*)(ws + 163840);   // 24*4  tiles -> 96 KB
  _Float16* pk_ebw1 = (_Float16*)(ws + 262144);   // 16*17 tiles -> 272 KB
  _Float16* pk_ebw2 = (_Float16*)(ws + 540672);   //  8*8  tiles -> 64 KB
  float*    pooled  = (float*)(ws + 606208);      // 128 floats
  _Float16* o3h     = (_Float16*)(ws + 655360);   // 200000*128 f16 = 51.2 MB
  float*    b2in    = (float*)(ws + 51855360);    // 8192*544 f32   = 17.8 MB

  // 0) repack weights into WMMA B-fragment layout (f16), zero pooled accum
  pack_weights<<<cdiv(8 * 8 * 32, 256), 256, 0, stream>>>(eo_w2, pk_eow2, 128, 256, 8, 8);
  pack_weights<<<cdiv(24 * 4 * 32, 256), 256, 0, stream>>>(gru_wih, pk_wih, 384, 128, 24, 4);
  pack_weights<<<cdiv(24 * 4 * 32, 256), 256, 0, stream>>>(gru_whh, pk_whh, 384, 128, 24, 4);
  pack_weights<<<cdiv(16 * 17 * 32, 256), 256, 0, stream>>>(eb_w1, pk_ebw1, 256, DIM_B, 16, 17);
  pack_weights<<<cdiv(8 * 8 * 32, 256), 256, 0, stream>>>(eb_w2, pk_ebw2, 128, 256, 8, 8);
  zero128<<<1, 128, 0, stream>>>(pooled);

  // 1) order embedding (200000 orders, 128/block)
  order_embed<<<cdiv(N_ORDERS, 128), 256, 0, stream>>>(
      order_usage, ln_order_g, ln_order_b, eo_w1, eo_b1, pk_eow2, eo_b2, o3h);

  // 2) GRU + pooling (8192 rows, 32/block)
  gru_kernel<<<N_BATCH / GM, 256, 0, stream>>>(
      o3h, plan_idx, plan_len, pk_wih, pk_whh, gru_bih, gru_bhh,
      hist, plan_usage, b2in);

  // 3) TDM-staged LN + MLP head + pooled accumulation
  head_kernel<<<N_BATCH / GM, 256, 0, stream>>>(
      b2in, lnb_g, lnb_b, pk_ebw1, eb_b1, pk_ebw2, eb_b2, pooled);

  // 4) scalar value head
  final_kernel<<<1, 128, 0, stream>>>(pooled, cr_w1, cr_b1, cr_w2, cr_b2,
                                      (float*)d_out);
}
